// InfoGraph_2619930051185
// MI455X (gfx1250) — matrix operand order repaired and checked by reading, not compiled, not run on hardware
//
#include <hip/hip_runtime.h>
#include <hip/hip_bf16.h>

typedef __attribute__((ext_vector_type(16))) __bf16        v16bf;
typedef __attribute__((ext_vector_type(8)))  float         v8f;
typedef __attribute__((ext_vector_type(4)))  unsigned int  v4u;

#define DD 512
#define NN 50000
#define GG 1024
#define LDB 520              /* padded LDS row pitch (bf16 elems) -> 1040 B, bank-conflict-free */
#define LOG2_F 0.69314718055994530942f

union FragU { v16bf v; v4u q[2]; };

// A-matrix 16x32 bf16 fragment (lane = M row, half-wave selects K-subgroups):
// elements 0..7  <- K = k0 + 8*hi .. +7 ; elements 8..15 <- K = k0 + 16 + 8*hi .. +7
__device__ inline v16bf load_a_frag(const __hip_bfloat16* rowbase, int k0, int hi) {
    FragU f;
    const v4u* p = reinterpret_cast<const v4u*>(rowbase + k0 + hi * 8);
    f.q[0] = p[0];
    f.q[1] = p[2];
    return f.v;
}

// B-matrix 32x16 bf16 fragment from LDS (lane = N col, same K groups as A: k0+8*hi..+7, k0+16+8*hi..+7)
__device__ inline v16bf load_b_frag_lds(const __hip_bfloat16* Bs, int n, int k0, int hi) {
    FragU f;
    const v4u* p = reinterpret_cast<const v4u*>(Bs + (size_t)n * LDB + k0 + hi * 8);
    f.q[0] = p[0];
    f.q[1] = p[2];
    return f.v;
}

// Async-stage 64 contiguous rows (64 x 512 bf16 = 64 KB) from global into LDS with
// padded row pitch.  CDNA5 path: global_load_async_to_lds_b128 (ASYNCcnt tracked).
__device__ inline void stage64rows_async(const __hip_bfloat16* __restrict__ src,
                                         __hip_bfloat16* Bs) {
    const unsigned lds_base = (unsigned)(uintptr_t)Bs;   // low 32 bits of generic ptr = LDS offset
    const int t = threadIdx.x;                            // 0..255
#pragma unroll
    for (int i = 0; i < 16; ++i) {
        const int c = t + i * 256;                        // 16-byte chunk id, 0..4095
        const unsigned lds_off = lds_base + (unsigned)((c >> 5) * (LDB * 2) + (c & 31) * 16);
        const unsigned goff = (unsigned)c * 16u;
        asm volatile("global_load_async_to_lds_b128 %0, %1, %2"
                     :: "v"(lds_off), "v"(goff), "s"(src) : "memory");
    }
    asm volatile("s_wait_asynccnt 0x0" ::: "memory");
}

__device__ inline float softplusf(float x) {
    return fmaxf(x, 0.0f) + log1pf(__expf(-fabsf(x)));   // numerically stable
}

__global__ __launch_bounds__(256)
void cvt_f32_bf16(const float* __restrict__ in, __hip_bfloat16* __restrict__ out, long long n) {
    long long i = (long long)blockIdx.x * blockDim.x + threadIdx.x;
    long long stride = (long long)gridDim.x * blockDim.x;
    for (; i < n; i += stride) out[i] = __float2bfloat16(in[i]);
}

// Out[M,512] = act(A[M,512] @ W[512,512]^T + b) [+ Skip]
// Workgroup: 128(M) x 64(N); weight tile staged once in LDS, shared by 8 waves.
template<bool RELU, bool SKIP>
__global__ __launch_bounds__(256)
void ff_gemm(const __hip_bfloat16* __restrict__ A,
             const __hip_bfloat16* __restrict__ W,
             const float* __restrict__ bias,
             const __hip_bfloat16* __restrict__ Skip,
             __hip_bfloat16* __restrict__ Out,
             int M)
{
    __shared__ __hip_bfloat16 Bs[64 * LDB];

    const int colBlocks = DD / 64;                 // 8
    const int mblk = blockIdx.x / colBlocks;
    const int n0   = (blockIdx.x % colBlocks) * 64;

    stage64rows_async(W + (size_t)n0 * DD, Bs);    // rows n0..n0+63 are contiguous
    __syncthreads();

    const int lane = threadIdx.x & 31;
    const int wave = threadIdx.x >> 5;
    const int Mt   = M >> 4;
    const int mt   = mblk * 8 + wave;
    if (mt < Mt) {                                 // wave-uniform guard (EXEC all ones inside)
        const int row = lane & 15;
        const int hi  = lane >> 4;
        const __hip_bfloat16* arow = A + (size_t)(mt * 16 + row) * DD;

        v8f acc[4] = { {}, {}, {}, {} };
#pragma unroll
        for (int k0 = 0; k0 < DD; k0 += 32) {
            v16bf a = load_a_frag(arow, k0, hi);
#pragma unroll
            for (int j = 0; j < 4; ++j) {
                v16bf b = load_b_frag_lds(Bs, j * 16 + row, k0, hi);
                acc[j] = __builtin_amdgcn_wmma_f32_16x16x32_bf16(
                    false, a, false, b, (short)0, acc[j], false, false);
            }
        }

#pragma unroll
        for (int j = 0; j < 4; ++j) {
            const int n = n0 + j * 16 + row;
            const float bn = bias[n];
#pragma unroll
            for (int v = 0; v < 8; ++v) {
                const int m = mt * 16 + v + hi * 8;
                float val = acc[j][v] + bn;
                if (RELU) val = fmaxf(val, 0.0f);
                if (SKIP) val += __bfloat162float(Skip[(size_t)m * DD + n]);
                Out[(size_t)m * DD + n] = __float2bfloat16(val);
            }
        }
    }
}

// res tile = L[128,512] @ G[64,512]^T with g_enc tile LDS-staged; fused JSD reduction.
__global__ __launch_bounds__(256)
void loss_kernel(const __hip_bfloat16* __restrict__ L,
                 const __hip_bfloat16* __restrict__ Gm,
                 const int* __restrict__ batch,
                 float* __restrict__ accum)
{
    __shared__ __hip_bfloat16 Bs[64 * LDB];

    const int colBlocks = GG / 64;                 // 16
    const int mblk = blockIdx.x / colBlocks;
    const int g0   = (blockIdx.x % colBlocks) * 64;

    stage64rows_async(Gm + (size_t)g0 * DD, Bs);
    __syncthreads();

    const int lane = threadIdx.x & 31;
    const int wave = threadIdx.x >> 5;
    const int It   = NN / 16;                      // 3125
    const int it   = mblk * 8 + wave;

    float pos_s = 0.0f, neg_s = 0.0f;
    if (it < It) {                                 // wave-uniform guard
        const int row = lane & 15;
        const int hi  = lane >> 4;
        const __hip_bfloat16* lrow = L + (size_t)(it * 16 + row) * DD;

        v8f acc[4] = { {}, {}, {}, {} };
#pragma unroll
        for (int k0 = 0; k0 < DD; k0 += 32) {
            v16bf a = load_a_frag(lrow, k0, hi);
#pragma unroll
            for (int j = 0; j < 4; ++j) {
                v16bf b = load_b_frag_lds(Bs, j * 16 + row, k0, hi);
                acc[j] = __builtin_amdgcn_wmma_f32_16x16x32_bf16(
                    false, a, false, b, (short)0, acc[j], false, false);
            }
        }

        int bidx[8];
#pragma unroll
        for (int v = 0; v < 8; ++v) bidx[v] = batch[it * 16 + v + hi * 8];

#pragma unroll
        for (int j = 0; j < 4; ++j) {
            const int g = g0 + j * 16 + row;
#pragma unroll
            for (int v = 0; v < 8; ++v) {
                const float r  = acc[j][v];
                const float sp = softplusf(-r);
                if (bidx[v] == g) pos_s += (LOG2_F - sp);      // E_pos term
                else              neg_s += (sp + r - LOG2_F);  // E_neg term
            }
        }
    }

#pragma unroll
    for (int off = 16; off > 0; off >>= 1) {        // wave32 reduction
        pos_s += __shfl_xor(pos_s, off, 32);
        neg_s += __shfl_xor(neg_s, off, 32);
    }
    if ((threadIdx.x & 31) == 0) {
        atomicAdd(&accum[0], pos_s);
        atomicAdd(&accum[1], neg_s);
    }
}

__global__ void finalize_kernel(const float* __restrict__ accum, float* __restrict__ out) {
    const float E_pos = accum[0] / (float)NN;
    const float E_neg = accum[1] / ((float)NN * (float)(GG - 1));
    out[0] = E_neg - E_pos;
}

extern "C" void kernel_launch(void* const* d_in, const int* in_sizes, int n_in,
                              void* d_out, int out_size, void* d_ws, size_t ws_size,
                              hipStream_t stream)
{
    (void)in_sizes; (void)n_in; (void)out_size; (void)ws_size;

    const float* node_rep  = (const float*)d_in[0];
    const float* graph_rep = (const float*)d_in[1];
    const int*   batch     = (const int*)d_in[2];
    const float* Wf[8] = { (const float*)d_in[3],  (const float*)d_in[5],
                           (const float*)d_in[7],  (const float*)d_in[9],
                           (const float*)d_in[11], (const float*)d_in[13],
                           (const float*)d_in[15], (const float*)d_in[17] };
    const float* Bf[8] = { (const float*)d_in[4],  (const float*)d_in[6],
                           (const float*)d_in[8],  (const float*)d_in[10],
                           (const float*)d_in[12], (const float*)d_in[14],
                           (const float*)d_in[16], (const float*)d_in[18] };

    char* ws = (char*)d_ws;
    float* accum = (float*)ws;
    size_t off = 256;
    const size_t NB = (size_t)NN * DD * sizeof(__hip_bfloat16);   // 51.2 MB
    const size_t GB = (size_t)GG * DD * sizeof(__hip_bfloat16);   // 1 MB
    const size_t WB = (size_t)DD * DD * sizeof(__hip_bfloat16);   // 0.5 MB
    __hip_bfloat16* Xn   = (__hip_bfloat16*)(ws + off); off += NB;
    __hip_bfloat16* BufA = (__hip_bfloat16*)(ws + off); off += NB;
    __hip_bfloat16* BufB = (__hip_bfloat16*)(ws + off); off += NB;
    __hip_bfloat16* Xg   = (__hip_bfloat16*)(ws + off); off += GB;
    __hip_bfloat16* Hg1  = (__hip_bfloat16*)(ws + off); off += GB;
    __hip_bfloat16* Hg2  = (__hip_bfloat16*)(ws + off); off += GB;
    __hip_bfloat16* Genc = (__hip_bfloat16*)(ws + off); off += GB;
    __hip_bfloat16* Wb[8];
    for (int i = 0; i < 8; ++i) { Wb[i] = (__hip_bfloat16*)(ws + off); off += WB; }

    hipMemsetAsync(accum, 0, 2 * sizeof(float), stream);

    cvt_f32_bf16<<<4096, 256, 0, stream>>>(node_rep,  Xn, (long long)NN * DD);
    cvt_f32_bf16<<<512,  256, 0, stream>>>(graph_rep, Xg, (long long)GG * DD);
    for (int i = 0; i < 8; ++i)
        cvt_f32_bf16<<<256, 256, 0, stream>>>(Wf[i], Wb[i], (long long)DD * DD);

    const int nRowBlk = (NN / 16 + 7) / 8;   // 391
    const int gRowBlk = (GG / 16 + 7) / 8;   // 8

    // node FF: 3x relu-linear + linear shortcut
    ff_gemm<true,  false><<<nRowBlk * 8, 256, 0, stream>>>(Xn,   Wb[0], Bf[0], nullptr, BufA, NN);
    ff_gemm<true,  false><<<nRowBlk * 8, 256, 0, stream>>>(BufA, Wb[1], Bf[1], nullptr, BufB, NN);
    ff_gemm<true,  false><<<nRowBlk * 8, 256, 0, stream>>>(BufB, Wb[2], Bf[2], nullptr, BufA, NN);
    ff_gemm<false, true ><<<nRowBlk * 8, 256, 0, stream>>>(Xn,   Wb[3], Bf[3], BufA,    BufB, NN);

    // graph FF
    ff_gemm<true,  false><<<gRowBlk * 8, 256, 0, stream>>>(Xg,  Wb[4], Bf[4], nullptr, Hg1,  GG);
    ff_gemm<true,  false><<<gRowBlk * 8, 256, 0, stream>>>(Hg1, Wb[5], Bf[5], nullptr, Hg2,  GG);
    ff_gemm<true,  false><<<gRowBlk * 8, 256, 0, stream>>>(Hg2, Wb[6], Bf[6], nullptr, Hg1,  GG);
    ff_gemm<false, true ><<<gRowBlk * 8, 256, 0, stream>>>(Xg,  Wb[7], Bf[7], Hg1,     Genc, GG);

    // fused res-GEMM + JSD reduction
    loss_kernel<<<nRowBlk * (GG / 64), 256, 0, stream>>>(BufB, Genc, batch, accum);
    finalize_kernel<<<1, 1, 0, stream>>>(accum, (float*)d_out);
}
